// MSDeformAttn_5162550690554
// MI455X (gfx1250) — matrix operand order repaired and checked by reading, not compiled, not run on hardware
//
#include <hip/hip_runtime.h>
#include <hip/hip_bf16.h>
#include <stdint.h>

// MSDeformAttn for MI455X (gfx1250, wave32, WMMA + TDM).
// B=2, C=256, HD=8, D=32, NL=4, NP=4, LQ=LV=21760.

#define BATCH 2
#define CDIM  256
#define HEADS 8
#define DHEAD 32
#define NLEV  4
#define NPTS  4
#define LQN   21760
#define LVN   21760
#define MROWS (BATCH * LQN)   // 43520

typedef __attribute__((ext_vector_type(16))) __bf16 v16bf;
typedef __attribute__((ext_vector_type(8)))  float  v8f;
typedef __attribute__((ext_vector_type(4)))  unsigned int u32x4;
typedef __attribute__((ext_vector_type(4)))  int          i32x4;
typedef __attribute__((ext_vector_type(8)))  int          i32x8;

__device__ __forceinline__ unsigned short f2bf(float f) {
  unsigned int u = __float_as_uint(f);
  u += 0x7FFFu + ((u >> 16) & 1u);          // round-to-nearest-even
  return (unsigned short)(u >> 16);
}

// ---------------- fp32 -> bf16 (flat) ----------------
__global__ void cvt_bf16_kernel(const float* __restrict__ src,
                                unsigned short* __restrict__ dst, int n) {
  int i = blockIdx.x * blockDim.x + threadIdx.x;
  if (i < n) dst[i] = f2bf(src[i]);
}

// ---------------- fp32 [K][N] -> bf16 [N][K] (weight transpose) ----------------
__global__ void cvtT_bf16_kernel(const float* __restrict__ src,
                                 unsigned short* __restrict__ dst, int K, int N) {
  int i = blockIdx.x * blockDim.x + threadIdx.x;
  if (i < K * N) {
    int k = i / N, n = i % N;
    dst[(size_t)n * K + k] = f2bf(src[i]);
  }
}

// ---------------- bf16 WMMA GEMM ----------------
// Block = 8 waves = 256x64 macro-tile. B-panel (64 cols x K=256, bf16, 32KB)
// is DMA'd into LDS once per block by the Tensor Data Mover; each wave then
// computes a 32x64 tile: 2 A-frags (global) x 4 B-frags (LDS) -> 8 WMMAs/K-step.
__global__ void __launch_bounds__(256)
gemm_bf16_wmma(const unsigned short* __restrict__ A,    // [M][K] bf16 row-major
               const unsigned short* __restrict__ Bt,   // [N][K] bf16 (W transposed)
               const float* __restrict__ bias,          // [N]
               const float* __restrict__ residual,      // [M][N] fp32 or null
               float* __restrict__ outF,                // [M][N] fp32 or null
               unsigned short* __restrict__ outBF,      // [M][N] bf16 or null
               int M, int N, int K) {
  __shared__ unsigned short Bsh[64 * 256];              // 32KB B panel

  const int lane = threadIdx.x & 31;
  const int wid  = threadIdx.x >> 5;
  const int colGroups = N >> 6;                          // N/64
  const int mblk = blockIdx.x / colGroups;
  const int cg   = blockIdx.x % colGroups;
  const int n0   = cg << 6;                              // 64-col panel base
  const int m0w  = (mblk << 8) + (wid << 5);             // this wave's 32-row base

  // ---- TDM: load Bt[n0 .. n0+63][0 .. K) into LDS (one descriptor, wave 0) ----
  if (wid == 0) {
    const unsigned long long gaddr =
        (unsigned long long)(uintptr_t)(Bt + (size_t)n0 * K);
    // LDS byte offset of Bsh: generic LDS address truncated to low 32 bits.
    const unsigned int lds_off = (unsigned int)(uintptr_t)(void*)Bsh;
    u32x4 g0;
    g0[0] = 1u;                                          // count=1, user mode
    g0[1] = lds_off;                                     // lds_addr (bytes)
    g0[2] = (unsigned int)(gaddr & 0xFFFFFFFFu);         // global_addr[31:0]
    g0[3] = (unsigned int)((gaddr >> 32) & 0x1FFFFFFu)   // global_addr[56:32]
          | (2u << 30);                                  // type = 2 ("image")
    i32x8 g1;
    g1[0] = (int)(1u << 16);                             // data_size=1 -> 2 bytes
    g1[1] = (int)((unsigned)K << 16);                    // tensor_dim0[15:0] @bits63:48
    g1[2] = (int)(64u << 16);                            // dim0 hi=0 | tensor_dim1=64
    g1[3] = (int)((unsigned)K << 16);                    // dim1 hi=0 | tile_dim0=K
    g1[4] = 64;                                          // tile_dim1=64, tile_dim2=0
    g1[5] = K;                                           // tensor_dim0_stride lo
    g1[6] = 0;                                           // stride hi / dim1_stride lo
    g1[7] = 0;
    i32x4 gz = {0, 0, 0, 0};                             // groups 2/3: dims unused
#if defined(__clang_major__) && (__clang_major__ >= 23)
    i32x8 gz8 = {0, 0, 0, 0, 0, 0, 0, 0};
    __builtin_amdgcn_tensor_load_to_lds(g0, g1, gz, gz, gz8, 0);
#else
    __builtin_amdgcn_tensor_load_to_lds(g0, g1, gz, gz, 0);
#endif
    __builtin_amdgcn_s_wait_tensorcnt(0);
  }
  __syncthreads();

  const int half = lane >> 4;                            // 0 or 1
  const int l16  = lane & 15;

  const unsigned int* Arow0 = (const unsigned int*)(A + (size_t)(m0w + l16) * K);
  const unsigned int* Arow1 = (const unsigned int*)(A + (size_t)(m0w + 16 + l16) * K);

  // Launder the LDS pointer: the TDM wrote Bsh behind the compiler's back, so
  // force the loads to materialize (asm is a may-writer and hides provenance).
  const unsigned int* BshU = (const unsigned int*)Bsh;   // [64][K/2] u32
  asm volatile("" : "+v"(BshU) : : "memory");
  BshU = (const unsigned int*)__builtin_assume_aligned(BshU, 16);

  v8f acc[2][4];
#pragma unroll
  for (int i = 0; i < 2; i++)
#pragma unroll
    for (int t = 0; t < 4; t++) acc[i][t] = (v8f){};

  for (int kk = 0; kk < K; kk += 32) {
    union { unsigned int u[8]; v16bf v; } fa0, fa1, fb[4];
    // A 16x32 bf16 layout: vgpr j<4 -> K=2j+8*half, j>=4 -> K=16+2(j-4)+8*half
    const int ab = (kk >> 1) + 4 * half;
#pragma unroll
    for (int j = 0; j < 8; j++) {
      const int ai = ab + j + ((j >= 4) ? 4 : 0);
      fa0.u[j] = Arow0[ai];
      fa1.u[j] = Arow1[ai];
    }
    // B 32x16 bf16 layout: same K order as A (vgpr j<4 -> K=2j+8*half, j>=4 -> K=16+2(j-4)+8*half)
    const int bb = (kk >> 1) + 4 * half;
#pragma unroll
    for (int t = 0; t < 4; t++) {
      const int nL = (t << 4) + l16;                     // local col in panel
#pragma unroll
      for (int j = 0; j < 8; j++) fb[t].u[j] = BshU[nL * (K >> 1) + bb + j + ((j >= 4) ? 4 : 0)];
    }
#pragma unroll
    for (int t = 0; t < 4; t++) {
      acc[0][t] = __builtin_amdgcn_wmma_f32_16x16x32_bf16(
          false, fa0.v, false, fb[t].v, (short)0, acc[0][t], false, false);
      acc[1][t] = __builtin_amdgcn_wmma_f32_16x16x32_bf16(
          false, fa1.v, false, fb[t].v, (short)0, acc[1][t], false, false);
    }
  }

  float bcol[4];
#pragma unroll
  for (int t = 0; t < 4; t++) bcol[t] = bias[n0 + (t << 4) + l16];

#pragma unroll
  for (int i = 0; i < 2; i++) {
#pragma unroll
    for (int v = 0; v < 8; v++) {
      const int r = m0w + (i << 4) + v + 8 * half;       // C/D layout
      const size_t rowoff = (size_t)r * N;
#pragma unroll
      for (int t = 0; t < 4; t++) {
        const size_t idx = rowoff + n0 + (t << 4) + l16;
        float val = acc[i][t][v] + bcol[t];
        if (residual) val += residual[idx];
        if (outF)  outF[idx]  = val;
        if (outBF) outBF[idx] = f2bf(val);
      }
    }
  }
}

// ---------------- softmax + bilinear sampling + head accumulation ----------------
// One wave per (b, q, h); lane = channel d in [0,32).
__global__ void __launch_bounds__(256)
msda_sample_kernel(const float* __restrict__ vproj,    // [B*LV][C] fp32
                   const float* __restrict__ offm,     // [B*LQ][256] fp32
                   const float* __restrict__ attn,     // [B*LQ][128] fp32 logits
                   const float* __restrict__ refp,     // [B][LQ][NL][2]
                   unsigned short* __restrict__ outhead) { // bf16 [B*LQ][C]
  const int lane = threadIdx.x & 31;
  const int wid  = threadIdx.x >> 5;
  const long w = (long)blockIdx.x * 8 + wid;
  const long totalW = (long)BATCH * LQN * HEADS;
  if (w >= totalW) return;                        // wave-uniform

  const int  h  = (int)(w % HEADS);
  const long bq = w / HEADS;                      // b*LQ + q
  const int  b  = (int)(bq / LQN);

  float lg[16];
  const float* ap = attn + bq * 128 + h * 16;
#pragma unroll
  for (int i = 0; i < 16; i++) lg[i] = ap[i];
  float mx = lg[0];
#pragma unroll
  for (int i = 1; i < 16; i++) mx = fmaxf(mx, lg[i]);
  float sum = 0.f;
#pragma unroll
  for (int i = 0; i < 16; i++) { lg[i] = __expf(lg[i] - mx); sum += lg[i]; }
  const float inv = 1.f / sum;

  const int Wl_[4] = {128, 64, 32, 16};
  const int Hl_[4] = {128, 64, 32, 16};
  const int st_[4] = {0, 16384, 20480, 21504};

  const float* op = offm + bq * 256 + h * (NLEV * NPTS * 2);
  const float* rp = refp + bq * (NLEV * 2);
  const int col = h * DHEAD + lane;
  float accv = 0.f;

#pragma unroll
  for (int l = 0; l < NLEV; l++) {
    const int Wl = Wl_[l], Hl = Hl_[l];
    const long rbase = (long)b * LVN + st_[l];
    const float rx = rp[l * 2 + 0], ry = rp[l * 2 + 1];
    const float invW = 1.f / (float)Wl, invH = 1.f / (float)Hl;
#pragma unroll
    for (int p = 0; p < NPTS; p++) {
      const float ox = op[(l * NPTS + p) * 2 + 0];
      const float oy = op[(l * NPTS + p) * 2 + 1];
      const float aw = lg[l * NPTS + p] * inv;
      const float x = (rx + ox * invW) * (float)Wl - 0.5f;
      const float y = (ry + oy * invH) * (float)Hl - 0.5f;
      const float xf = floorf(x), yf = floorf(y);
      const float lx = x - xf, ly = y - yf;
      const int x0 = (int)xf, y0 = (int)yf;
      float s = 0.f;
      { const int xi = x0,     yi = y0;     const float cw = (1.f - lx) * (1.f - ly);
        if (xi >= 0 && xi < Wl && yi >= 0 && yi < Hl)
          s += cw * vproj[(rbase + (long)yi * Wl + xi) * CDIM + col]; }
      { const int xi = x0 + 1, yi = y0;     const float cw = lx * (1.f - ly);
        if (xi >= 0 && xi < Wl && yi >= 0 && yi < Hl)
          s += cw * vproj[(rbase + (long)yi * Wl + xi) * CDIM + col]; }
      { const int xi = x0,     yi = y0 + 1; const float cw = (1.f - lx) * ly;
        if (xi >= 0 && xi < Wl && yi >= 0 && yi < Hl)
          s += cw * vproj[(rbase + (long)yi * Wl + xi) * CDIM + col]; }
      { const int xi = x0 + 1, yi = y0 + 1; const float cw = lx * ly;
        if (xi >= 0 && xi < Wl && yi >= 0 && yi < Hl)
          s += cw * vproj[(rbase + (long)yi * Wl + xi) * CDIM + col]; }
      accv += aw * s;
    }
  }
  outhead[bq * CDIM + col] = f2bf(accv);
}

extern "C" void kernel_launch(void* const* d_in, const int* in_sizes, int n_in,
                              void* d_out, int out_size, void* d_ws, size_t ws_size,
                              hipStream_t stream) {
  (void)in_sizes; (void)n_in; (void)out_size; (void)ws_size;
  const float* query  = (const float*)d_in[0];   // [B,LQ,C]
  const float* refp   = (const float*)d_in[1];   // [B,LQ,NL,2]
  const float* value  = (const float*)d_in[2];   // [B,LV,C]
  // d_in[3] spatial_shapes (i64), d_in[4] level_start_index (i64): static, unused
  const float* W_off  = (const float*)d_in[5];   // [C,256]
  const float* b_off  = (const float*)d_in[6];
  const float* W_attn = (const float*)d_in[7];   // [C,128]
  const float* b_attn = (const float*)d_in[8];
  const float* W_val  = (const float*)d_in[9];   // [C,C]
  const float* b_val  = (const float*)d_in[10];
  const float* W_out  = (const float*)d_in[11];  // [C,C]
  const float* b_out  = (const float*)d_in[12];
  float* out = (float*)d_out;                    // [B,LQ,C] fp32

  const int M = MROWS;                           // 43520
  char* ws = (char*)d_ws;
  size_t o = 0;
  auto alloc = [&](size_t bytes) -> char* {
    char* p = ws + o;
    o = (o + bytes + 255) & ~(size_t)255;
    return p;
  };

  unsigned short* q_bf    = (unsigned short*)alloc((size_t)M * CDIM * 2);
  unsigned short* v_bf    = (unsigned short*)alloc((size_t)M * CDIM * 2);
  unsigned short* WvalT   = (unsigned short*)alloc((size_t)CDIM * CDIM * 2);
  unsigned short* WoffT   = (unsigned short*)alloc((size_t)CDIM * 256 * 2);
  unsigned short* WattnT  = (unsigned short*)alloc((size_t)CDIM * 128 * 2);
  unsigned short* WoutT   = (unsigned short*)alloc((size_t)CDIM * CDIM * 2);
  float*          vproj   = (float*)alloc((size_t)M * CDIM * 4);
  float*          offm    = (float*)alloc((size_t)M * 256 * 4);
  float*          attnw   = (float*)alloc((size_t)M * 128 * 4);
  unsigned short* outhead = (unsigned short*)alloc((size_t)M * CDIM * 2);

  const int nAct = M * CDIM;                      // 11,141,120
  cvt_bf16_kernel<<<(nAct + 255) / 256, 256, 0, stream>>>(query, q_bf, nAct);
  cvt_bf16_kernel<<<(nAct + 255) / 256, 256, 0, stream>>>(value, v_bf, nAct);
  cvtT_bf16_kernel<<<(CDIM * 256 + 255) / 256, 256, 0, stream>>>(W_val,  WvalT,  CDIM, 256);
  cvtT_bf16_kernel<<<(CDIM * 256 + 255) / 256, 256, 0, stream>>>(W_off,  WoffT,  CDIM, 256);
  cvtT_bf16_kernel<<<(CDIM * 128 + 255) / 256, 256, 0, stream>>>(W_attn, WattnT, CDIM, 128);
  cvtT_bf16_kernel<<<(CDIM * 256 + 255) / 256, 256, 0, stream>>>(W_out,  WoutT,  CDIM, 256);

  // grid: (M/256 row-blocks) x (N/64 col-panels)
  auto gemm_blocks = [&](int n) { return (M >> 8) * (n >> 6); };

  // v = value @ W_val + b_val  -> fp32 (consumed by gather)
  gemm_bf16_wmma<<<gemm_blocks(256), 256, 0, stream>>>(
      v_bf, WvalT, b_val, nullptr, vproj, nullptr, M, 256, CDIM);
  // off = query @ W_off + b_off
  gemm_bf16_wmma<<<gemm_blocks(256), 256, 0, stream>>>(
      q_bf, WoffT, b_off, nullptr, offm, nullptr, M, 256, CDIM);
  // attn logits = query @ W_attn + b_attn
  gemm_bf16_wmma<<<gemm_blocks(128), 256, 0, stream>>>(
      q_bf, WattnT, b_attn, nullptr, attnw, nullptr, M, 128, CDIM);

  // softmax + bilinear sampling + head concat (bf16 for the final GEMM)
  const long waves = (long)BATCH * LQN * HEADS;   // 348160
  msda_sample_kernel<<<(int)((waves + 7) / 8), 256, 0, stream>>>(
      vproj, offm, attnw, refp, outhead);

  // out = outhead @ W_out + b_out + query   (residual fused)
  gemm_bf16_wmma<<<gemm_blocks(256), 256, 0, stream>>>(
      outhead, WoutT, b_out, query, out, nullptr, M, 256, CDIM);
}
